// GCN_MLC_63780264346284
// MI455X (gfx1250) — matrix operand order repaired and checked by reading, not compiled, not run on hardware
//
#include <hip/hip_runtime.h>

typedef _Float16 v16h __attribute__((ext_vector_type(16)));
typedef float    v8f  __attribute__((ext_vector_type(8)));
typedef float    v4f  __attribute__((ext_vector_type(4)));

#define FEAT 512
#define H1   32
#define H2   64

// ---------------- degree / CSR build ----------------

__global__ void k_init_deg(unsigned int* deg, int N) {
  int i = blockIdx.x * blockDim.x + threadIdx.x;
  if (i < N) deg[i] = 0u;
}

__global__ void k_count(const long long* __restrict__ ei, unsigned int* deg, int E) {
  int e = blockIdx.x * blockDim.x + threadIdx.x;
  if (e < E) {
    int d = (int)ei[(size_t)E + e];   // dst row
    atomicAdd(&deg[d], 1u);
  }
}

// ---- hierarchical exclusive scan (barrier-light, wave32 shfl-based) ----
// stage 1: per-block (256 elems) scan via wave shuffles; block totals -> partials
__global__ void __launch_bounds__(256) k_scan_blocks(const unsigned int* __restrict__ deg,
                                                     int* cursor, float* dinv,
                                                     int* partials, int N) {
  int tid = threadIdx.x;
  int i = blockIdx.x * 256 + tid;
  int lane = tid & 31, w = tid >> 5;
  int v = (i < N) ? (int)deg[i] : 0;
  int s = v;
#pragma unroll
  for (int off = 1; off < 32; off <<= 1) {       // wave-inclusive scan, no barriers
    int t = __shfl_up(s, off, 32);
    if (lane >= off) s += t;
  }
  __shared__ int wsum[8];
  if (lane == 31) wsum[w] = s;
  __syncthreads();
  int wexcl = 0;
#pragma unroll
  for (int k = 0; k < 8; ++k)
    if (k < w) wexcl += wsum[k];
  int excl = wexcl + s - v;
  if (i < N) {
    cursor[i] = excl;                            // block-local exclusive prefix
    dinv[i]   = rsqrtf((float)v + 1.0f);         // deg incl. self-loop
  }
  if (tid == 255) partials[blockIdx.x] = wexcl + s;  // block total
}

// stage 2: single 512-wide scan of block partials (<=512 blocks)
__global__ void k_scan_partials(int* partials, int nB) {
  __shared__ int sb[512];
  int tid = threadIdx.x;
  int v = (tid < nB) ? partials[tid] : 0;
  sb[tid] = v;
  __syncthreads();
  for (int off = 1; off < 512; off <<= 1) {
    int t = (tid >= off) ? sb[tid - off] : 0;
    __syncthreads();
    sb[tid] += t;
    __syncthreads();
  }
  if (tid < nB) partials[tid] = sb[tid] - v;     // exclusive block offsets
}

// stage 3: add block offsets
__global__ void k_scan_add(int* cursor, const int* __restrict__ partials, int N) {
  int i = blockIdx.x * 256 + threadIdx.x;
  if (i < N) cursor[i] += partials[blockIdx.x];
}

__global__ void k_fill(const long long* __restrict__ ei, int* cursor,
                       int* col, int E) {
  int e = blockIdx.x * blockDim.x + threadIdx.x;
  if (e < E) {
    int s = (int)ei[e];
    int d = (int)ei[(size_t)E + e];
    int p = atomicAdd(&cursor[d], 1);
    col[p] = s;
  }
}

// ---------------- weight packing into WMMA B-register layout ----------------
// B (K x 16) per v_wmma_f32_16x16x32_f16: lane = hi*16 + n, half j -> K = 8*hi + j + 8*(j>=8).
// Layout: W1p[((kb*2 + ntile)*32 + lane)*16 + j]

__global__ void k_packW1(const float* __restrict__ W1, _Float16* W1p) {
  int idx = blockIdx.x * blockDim.x + threadIdx.x;   // 16 kb * 2 nt * 32 lanes * 16 j
  int j    = idx & 15;
  int lane = (idx >> 4) & 31;
  int nt   = (idx >> 9) & 1;
  int kb   = idx >> 10;
  int hi = lane >> 4, l16 = lane & 15;
  int K = kb * 32 + 8 * hi + j + ((j >> 3) << 3);
  W1p[idx] = (_Float16)W1[K * H1 + nt * 16 + l16];
}

__global__ void k_packW2(const float* __restrict__ W2, _Float16* W2p) {
  int idx = blockIdx.x * blockDim.x + threadIdx.x;   // 4 nt * 32 lanes * 16 j
  int j    = idx & 15;
  int lane = (idx >> 4) & 31;
  int nt   = idx >> 9;
  int hi = lane >> 4, l16 = lane & 15;
  int K = 8 * hi + j + ((j >> 3) << 3);
  W2p[idx] = (_Float16)W2[K * H2 + nt * 16 + l16];
}

// ---------------- GEMM1: xws = dinv[m] * (x @ W1), f16 WMMA / f32 acc ----------------
// A (16 x K) per-lane layout: row m = lane&15, half j -> K = 8*hi + j + 8*(j>=8)
// => two contiguous 8-float segments per k-step: [kb+8*hi, +8) and [kb+16+8*hi, +8)

__global__ void __launch_bounds__(256) k_gemm1(const float* __restrict__ x,
                                               const _Float16* __restrict__ W1p,
                                               const float* __restrict__ dinv,
                                               float* __restrict__ xws, int nTiles) {
  int wave = (blockIdx.x * blockDim.x + threadIdx.x) >> 5;
  if (wave >= nTiles) return;                       // wave-uniform: EXEC stays full
  int lane = threadIdx.x & 31;
  int hi = lane >> 4, l16 = lane & 15;
  int rowbase = wave * 16;
  const float* xrow = x + (size_t)(rowbase + l16) * FEAT;

  v8f c0 = {0.f,0.f,0.f,0.f,0.f,0.f,0.f,0.f};
  v8f c1 = {0.f,0.f,0.f,0.f,0.f,0.f,0.f,0.f};

  for (int kb = 0; kb < FEAT; kb += 32) {
    const v4f* p0 = (const v4f*)(xrow + kb + 8 * hi);
    v4f s0 = p0[0], s1 = p0[1];
    const v4f* p1 = (const v4f*)(xrow + kb + 16 + 8 * hi);
    v4f s2 = p1[0], s3 = p1[1];
    v16h a;
    a[0]=(_Float16)s0[0]; a[1]=(_Float16)s0[1]; a[2]=(_Float16)s0[2]; a[3]=(_Float16)s0[3];
    a[4]=(_Float16)s1[0]; a[5]=(_Float16)s1[1]; a[6]=(_Float16)s1[2]; a[7]=(_Float16)s1[3];
    a[8]=(_Float16)s2[0]; a[9]=(_Float16)s2[1]; a[10]=(_Float16)s2[2]; a[11]=(_Float16)s2[3];
    a[12]=(_Float16)s3[0]; a[13]=(_Float16)s3[1]; a[14]=(_Float16)s3[2]; a[15]=(_Float16)s3[3];

    const v16h* bp = (const v16h*)(W1p + (size_t)(kb >> 5) * 1024 + lane * 16);
    v16h b0 = bp[0];      // ntile 0
    v16h b1 = bp[32];     // ntile 1 (+512 halves)

    c0 = __builtin_amdgcn_wmma_f32_16x16x32_f16(false, a, false, b0, (short)0, c0, false, false);
    c1 = __builtin_amdgcn_wmma_f32_16x16x32_f16(false, a, false, b1, (short)0, c1, false, false);
  }

  // C/D layout: VGPR r, lane -> row rowbase + r + 8*hi, col l16
#pragma unroll
  for (int r = 0; r < 8; ++r) {
    int mm = rowbase + r + 8 * hi;
    float dv = dinv[mm];
    xws[(size_t)mm * H1 + l16]      = c0[r] * dv;
    xws[(size_t)mm * H1 + 16 + l16] = c1[r] * dv;
  }
}

// ---------------- CSR gather aggregation + bias + ReLU ----------------
// one wave per node, lane = channel; per edge: 32-lane coalesced 128B L2-resident read.
// 4-way unrolled with independent accumulators to keep 4 gathers in flight.

__global__ void __launch_bounds__(256) k_aggregate(const float* __restrict__ xws,
                                                   const int* __restrict__ col,
                                                   const int* __restrict__ cursor,
                                                   const unsigned int* __restrict__ deg,
                                                   const float* __restrict__ dinv,
                                                   const float* __restrict__ b1,
                                                   float* __restrict__ hbuf, int N) {
  int node = (blockIdx.x * blockDim.x + threadIdx.x) >> 5;
  if (node >= N) return;
  int c = threadIdx.x & 31;
  int end = cursor[node];           // cursor advanced to row end by k_fill
  int cnt = (int)deg[node];
  int start = end - cnt;
  float s0 = xws[(size_t)node * H1 + c];   // self-loop term
  float s1 = 0.f, s2 = 0.f, s3 = 0.f;
  int p = start;
  for (; p + 4 <= end; p += 4) {
    int a0 = col[p], a1 = col[p + 1], a2 = col[p + 2], a3 = col[p + 3];
    s0 += xws[(size_t)a0 * H1 + c];
    s1 += xws[(size_t)a1 * H1 + c];
    s2 += xws[(size_t)a2 * H1 + c];
    s3 += xws[(size_t)a3 * H1 + c];
  }
  for (; p < end; ++p) {
    int a0 = col[p];
    s0 += xws[(size_t)a0 * H1 + c];
  }
  float s = (s0 + s1) + (s2 + s3);
  float h = dinv[node] * s + b1[c];
  hbuf[(size_t)node * H1 + c] = h > 0.f ? h : 0.f;
}

// ---------------- GEMM2: out = h @ W2 + b2, single-K WMMA ----------------

__global__ void __launch_bounds__(256) k_gemm2(const float* __restrict__ hbuf,
                                               const _Float16* __restrict__ W2p,
                                               const float* __restrict__ b2,
                                               float* __restrict__ out, int nTiles) {
  int wave = (blockIdx.x * blockDim.x + threadIdx.x) >> 5;
  if (wave >= nTiles) return;
  int lane = threadIdx.x & 31;
  int hi = lane >> 4, l16 = lane & 15;
  int rowbase = wave * 16;

  const float* hrow = hbuf + (size_t)(rowbase + l16) * H1;
  const v4f* p0 = (const v4f*)(hrow + 8 * hi);
  v4f s0 = p0[0], s1 = p0[1];
  const v4f* p1 = (const v4f*)(hrow + 16 + 8 * hi);
  v4f s2 = p1[0], s3 = p1[1];
  v16h a;
  a[0]=(_Float16)s0[0]; a[1]=(_Float16)s0[1]; a[2]=(_Float16)s0[2]; a[3]=(_Float16)s0[3];
  a[4]=(_Float16)s1[0]; a[5]=(_Float16)s1[1]; a[6]=(_Float16)s1[2]; a[7]=(_Float16)s1[3];
  a[8]=(_Float16)s2[0]; a[9]=(_Float16)s2[1]; a[10]=(_Float16)s2[2]; a[11]=(_Float16)s2[3];
  a[12]=(_Float16)s3[0]; a[13]=(_Float16)s3[1]; a[14]=(_Float16)s3[2]; a[15]=(_Float16)s3[3];

  v16h b0 = *(const v16h*)(W2p +    0 + lane * 16);
  v16h b1 = *(const v16h*)(W2p +  512 + lane * 16);
  v16h b2t= *(const v16h*)(W2p + 1024 + lane * 16);
  v16h b3 = *(const v16h*)(W2p + 1536 + lane * 16);

  v8f z = {0.f,0.f,0.f,0.f,0.f,0.f,0.f,0.f};
  v8f acc0 = __builtin_amdgcn_wmma_f32_16x16x32_f16(false, a, false, b0, (short)0, z, false, false);
  v8f acc1 = __builtin_amdgcn_wmma_f32_16x16x32_f16(false, a, false, b1, (short)0, z, false, false);
  v8f acc2 = __builtin_amdgcn_wmma_f32_16x16x32_f16(false, a, false, b2t,(short)0, z, false, false);
  v8f acc3 = __builtin_amdgcn_wmma_f32_16x16x32_f16(false, a, false, b3, (short)0, z, false, false);

  float bb0 = b2[l16], bb1 = b2[16 + l16], bb2 = b2[32 + l16], bb3 = b2[48 + l16];
#pragma unroll
  for (int r = 0; r < 8; ++r) {
    int m = rowbase + r + 8 * hi;
    float* o = out + (size_t)m * H2;
    o[l16]      = acc0[r] + bb0;
    o[16 + l16] = acc1[r] + bb1;
    o[32 + l16] = acc2[r] + bb2;
    o[48 + l16] = acc3[r] + bb3;
  }
}

// ---------------- launch ----------------

extern "C" void kernel_launch(void* const* d_in, const int* in_sizes, int n_in,
                              void* d_out, int out_size, void* d_ws, size_t ws_size,
                              hipStream_t stream) {
  const float*     x  = (const float*)d_in[0];
  const long long* ei = (const long long*)d_in[1];
  const float*     W1 = (const float*)d_in[2];
  const float*     b1 = (const float*)d_in[3];
  const float*     W2 = (const float*)d_in[4];
  const float*     b2 = (const float*)d_in[5];
  float* out = (float*)d_out;

  int N = in_sizes[0] / FEAT;   // 100000
  int E = in_sizes[1] / 2;      // 3200000

  char* w = (char*)d_ws;
  size_t off = 0;
  auto carve = [&](size_t bytes) -> void* {
    void* p = w + off;
    off += (bytes + 255) & ~(size_t)255;
    return p;
  };
  unsigned int* deg      = (unsigned int*)carve((size_t)N * 4);
  float*        dinv     = (float*)       carve((size_t)N * 4);
  int*          cursor   = (int*)         carve((size_t)N * 4);
  int*          partials = (int*)         carve(512 * 4);
  int*          col      = (int*)         carve((size_t)E * 4);
  float*        xws      = (float*)       carve((size_t)N * H1 * 4);
  float*        hbuf     = (float*)       carve((size_t)N * H1 * 4);
  _Float16*     W1p      = (_Float16*)    carve(16384 * 2);
  _Float16*     W2p      = (_Float16*)    carve(2048 * 2);
  (void)ws_size; (void)n_in; (void)out_size;

  int nBScan = (N + 255) / 256;             // 391 (<=512 for stage-2 scan)

  k_init_deg     <<<(N + 255) / 256, 256, 0, stream>>>(deg, N);
  k_count        <<<(E + 255) / 256, 256, 0, stream>>>(ei, deg, E);
  k_scan_blocks  <<<nBScan, 256, 0, stream>>>(deg, cursor, dinv, partials, N);
  k_scan_partials<<<1, 512, 0, stream>>>(partials, nBScan);
  k_scan_add     <<<nBScan, 256, 0, stream>>>(cursor, partials, N);
  k_fill         <<<(E + 255) / 256, 256, 0, stream>>>(ei, cursor, col, E);
  k_packW1       <<<16384 / 256, 256, 0, stream>>>(W1, W1p);
  k_packW2       <<<2048 / 256, 256, 0, stream>>>(W2, W2p);

  int nTiles = N / 16;                      // 6250 (N divisible by 16)
  int gGemm  = (nTiles + 7) / 8;            // 8 waves per 256-thread block
  k_gemm1    <<<gGemm, 256, 0, stream>>>(x, W1p, dinv, xws, nTiles);
  k_aggregate<<<(N + 7) / 8, 256, 0, stream>>>(xws, col, cursor, deg, dinv, b1, hbuf, N);
  k_gemm2    <<<gGemm, 256, 0, stream>>>(hbuf, W2p, b2, out, nTiles);
}
